// RNN_41910290874608
// MI455X (gfx1250) — matrix operand order repaired and checked by reading, not compiled, not run on hardware
//
#include <hip/hip_runtime.h>
#include <hip/hip_bf16.h>

// -----------------------------------------------------------------------------
// Bidirectional LiGRU stack for MI455X (gfx1250, wave32, WMMA).
//   B=8, T=2000, F=240, H=512, L=4.  2B=16 direction-rows per layer.
// Phase 1 (parallel): w[bb,t,g] = (xb @ W^T) * gamma' + beta   via bf16 WMMA.
// Phase 2 (serial):   h_{t+1} = sig(z)*h + (1-sig(z))*relu(a), gates = w_t + h U^T
//   single 32-wave workgroup, h in LDS (fp32 + bf16 shadow), U streamed from L2.
// Gate columns are PERMUTED so each wave's two WMMA tiles are the matched
// (a, z) pair for the same 16 hidden columns: p = pair*32 + [a:0..15 | z:16..31],
// orig g = (p&16) ? 512 + (p>>5)*16 + (p&15) : (p>>5)*16 + (p&15).
// -----------------------------------------------------------------------------

typedef __bf16 bf16;
typedef bf16  v16bf __attribute__((ext_vector_type(16)));
typedef float v8f   __attribute__((ext_vector_type(8)));
typedef unsigned int v4u __attribute__((ext_vector_type(4)));

#define T_LEN 2000
#define HDIM  512
#define GDIM  1024   // 2H
#define NB    16     // 2B direction-rows

union FragU { v16bf v; v4u q[2]; };

// A fragment (16x32 bf16): lane row m = lane%16; per lane 16 contiguous K values
// at k + (lane/16)*16.  `row` already points at this lane's row.
__device__ __forceinline__ v16bf load_a_16b(const bf16* row, int k, int lane) {
  const bf16* p = row + k + ((lane >> 4) << 4);
  FragU f;
  f.q[0] = *(const v4u*)p;
  f.q[1] = *(const v4u*)(p + 8);
  return f.v;
}

// B fragment (32x16 bf16), column n fixed per lane: 16 contiguous K values at
// k + (lane/16)*16.  `colrow` already points at this lane's column row (W[n]/U[n]).
__device__ __forceinline__ v16bf load_b_16b(const bf16* colrow, int k, int lane) {
  const bf16* p = colrow + k + ((lane >> 4) << 4);
  FragU f;
  f.q[0] = *(const v4u*)p;
  f.q[1] = *(const v4u*)(p + 8);
  return f.v;
}

__device__ __forceinline__ int inv_perm(int p) {
  int pair = p >> 5, within = p & 31;
  return (within < 16) ? (pair * 16 + within) : (512 + pair * 16 + (within - 16));
}

// ---------------- layer-0 input packing: x(8,T,240) -> xb bf16 (16,T,1024) ---
__global__ void ligru_pack_x(const float* __restrict__ x, bf16* __restrict__ xb) {
  const int bb = blockIdx.y;          // 0..15 (8 fwd, 8 bwd)
  const int t  = blockIdx.x;
  const int d  = threadIdx.x;         // 0..255  (240 real + 16 zero pad)
  const int b  = bb & 7;
  const int tt = (bb < 8) ? t : (T_LEN - 1 - t);
  float v = (d < 240) ? x[((size_t)b * T_LEN + tt) * 240 + d] : 0.0f;
  xb[((size_t)bb * T_LEN + t) * GDIM + d] = (bf16)v;
}

// ---------------- weight conversion (fp32 -> bf16, gate-permuted rows) -------
__global__ void ligru_conv_w(const float* __restrict__ W, bf16* __restrict__ Wb,
                             int Din, int Kp) {
  const int p = blockIdx.x;           // permuted gate row 0..1023
  const int g = inv_perm(p);
  for (int d = threadIdx.x; d < Kp; d += blockDim.x)
    Wb[(size_t)p * Kp + d] = (d < Din) ? (bf16)W[(size_t)g * Din + d] : (bf16)0.0f;
}

__global__ void ligru_conv_u(const float* __restrict__ U, bf16* __restrict__ Ub) {
  const int p = blockIdx.x;
  const int g = inv_perm(p);
  for (int k = threadIdx.x; k < HDIM; k += blockDim.x)
    Ub[(size_t)p * HDIM + k] = (bf16)U[(size_t)g * HDIM + k];
}

// ---------------- phase 1: w = xb @ W^T, fused BN scale/shift ----------------
// grid (T, 8), 256 threads = 8 waves; wave -> one 16-col tile; M tile = 16 rows
// (all direction-rows at one t).
__global__ __launch_bounds__(256)
void ligru_wx(const bf16* __restrict__ xb, const bf16* __restrict__ Wb,
              const float* __restrict__ gamma, const float* __restrict__ beta,
              float* __restrict__ wout, int Kp) {
  const int t    = blockIdx.x;
  const int lane = threadIdx.x & 31;
  const int wv   = threadIdx.x >> 5;
  const int n    = (blockIdx.y * 8 + wv) * 16 + (lane & 15);  // permuted gate col
  const int hi   = lane >> 4;
  const int g    = inv_perm(n);
  const float gs = gamma[g] * rsqrtf(1.0f + 1e-5f);
  const float bt = beta[g];
  const bf16* brow = Wb + (size_t)n * Kp;
  const bf16* arow = xb + ((size_t)(lane & 15) * T_LEN + t) * GDIM;
  v8f acc = {};
  for (int k = 0; k < Kp; k += 32) {
    v16bf a = load_a_16b(arow, k, lane);
    v16bf b = load_b_16b(brow, k, lane);
    acc = __builtin_amdgcn_wmma_f32_16x16x32_bf16(false, a, false, b,
                                                  (short)0, acc, false, false);
  }
#pragma unroll
  for (int r = 0; r < 8; ++r) {
    const int bb = hi * 8 + r;
    wout[((size_t)bb * T_LEN + t) * GDIM + n] = acc[r] * gs + bt;
  }
}

// ---------------- phase 2: serial recurrence, one 32-wave workgroup ----------
__global__ __launch_bounds__(1024)
void ligru_recurrence(const float* __restrict__ w, const bf16* __restrict__ Ub,
                      bf16* __restrict__ xb_next, float* __restrict__ out,
                      int last) {
  __shared__ float hF[NB * HDIM];     // 32 KB fp32 hidden state (accum precision)
  __shared__ bf16  hB[NB * HDIM];     // 16 KB bf16 shadow (WMMA A operand)
  const int tid  = threadIdx.x;
  const int lane = tid & 31;
  const int wv   = tid >> 5;          // 0..31, owns 16 hidden cols (a+z pair)
  const int c    = lane & 15;
  const int hi   = lane >> 4;
  const int j    = wv * 16 + c;       // hidden column owned by this lane

  for (int i = tid; i < NB * HDIM; i += 1024) { hF[i] = 0.0f; hB[i] = (bf16)0.0f; }
  __syncthreads();

  const bf16* u0 = Ub + (size_t)(wv * 32 + c) * HDIM;        // a-gate row
  const bf16* u1 = Ub + (size_t)(wv * 32 + 16 + c) * HDIM;   // z-gate row
  const bf16* hrow = hB + (lane & 15) * HDIM;                // A row m = lane%16

  for (int s = 0; s < T_LEN; ++s) {
    // accumulators initialized with precomputed w_t slice (gates = w_t + h U^T)
    v8f acc0, acc1;
#pragma unroll
    for (int r = 0; r < 8; ++r) {
      const int bb = hi * 8 + r;
      const float* wr = w + ((size_t)bb * T_LEN + s) * GDIM + wv * 32;
      acc0[r] = wr[c];
      acc1[r] = wr[16 + c];
    }
    // warm caches for next step's w slice while WMMAs run (global_prefetch_b8)
    if (s + 1 < T_LEN) {
#pragma unroll
      for (int r = 0; r < 8; ++r) {
        const int bb = hi * 8 + r;
        __builtin_prefetch(w + ((size_t)bb * T_LEN + s + 1) * GDIM + wv * 32, 0, 1);
      }
    }
#pragma unroll 4
    for (int k = 0; k < HDIM; k += 32) {
      v16bf a  = load_a_16b(hrow, k, lane);    // ds_load_b128 x2
      v16bf b0 = load_b_16b(u0, k, lane);      // global_load_b128 x2 (L2-resident U)
      v16bf b1 = load_b_16b(u1, k, lane);
      acc0 = __builtin_amdgcn_wmma_f32_16x16x32_bf16(false, a, false, b0,
                                                     (short)0, acc0, false, false);
      acc1 = __builtin_amdgcn_wmma_f32_16x16x32_bf16(false, a, false, b1,
                                                     (short)0, acc1, false, false);
    }
    __syncthreads();                  // everyone finished reading h_s

    // in-wave nonlinearity + state update for our (a,z) column pair
#pragma unroll
    for (int r = 0; r < 8; ++r) {
      const int bb = hi * 8 + r;
      const float at = acc0[r];
      const float zt = acc1[r];
      const float z  = 1.0f / (1.0f + __expf(-zt));
      const float hn = z * hF[bb * HDIM + j] + (1.0f - z) * fmaxf(at, 0.0f);
      const bf16 hb = (bf16)hn;
      hF[bb * HDIM + j] = hn;
      hB[bb * HDIM + j] = hb;
      if (last) {
        if (bb < 8)
          out[((size_t)bb * T_LEN + s) * GDIM + j] = hn;
        else
          out[((size_t)(bb - 8) * T_LEN + (T_LEN - 1 - s)) * GDIM + HDIM + j] = hn;
      } else {
        int d, r1, t1, r2, t2;
        if (bb < 8) { d = j;        r1 = bb;     t1 = s;              r2 = bb + 8; t2 = T_LEN - 1 - s; }
        else        { d = HDIM + j; r1 = bb - 8; t1 = T_LEN - 1 - s;  r2 = bb;     t2 = s; }
        xb_next[((size_t)r1 * T_LEN + t1) * GDIM + d] = hb;
        xb_next[((size_t)r2 * T_LEN + t2) * GDIM + d] = hb;
      }
    }
    __syncthreads();                  // h_{s+1} published
  }
}

// -----------------------------------------------------------------------------
extern "C" void kernel_launch(void* const* d_in, const int* in_sizes, int n_in,
                              void* d_out, int out_size, void* d_ws, size_t ws_size,
                              hipStream_t stream) {
  (void)in_sizes; (void)n_in; (void)out_size; (void)ws_size;
  const float* x = (const float*)d_in[0];
  char* ws = (char*)d_ws;
  const size_t XB = (size_t)NB * T_LEN * GDIM;       // elements per xb buffer
  bf16*  xbA  = (bf16*)(ws);                         // XB*2 bytes
  bf16*  xbB  = (bf16*)(ws + XB * 2);                // XB*2 bytes
  float* wbuf = (float*)(ws + XB * 4);               // XB*4 bytes
  bf16*  Wb   = (bf16*)(ws + XB * 8);                // 1024*1024*2 bytes
  bf16*  Ub   = (bf16*)(ws + XB * 8 + (size_t)GDIM * GDIM * 2); // 1024*512*2

  ligru_pack_x<<<dim3(T_LEN, NB), 256, 0, stream>>>(x, xbA);

  bf16* cur = xbA;
  bf16* nxt = xbB;
  for (int l = 0; l < 4; ++l) {
    const float* W     = (const float*)d_in[1 + 4 * l];
    const float* U     = (const float*)d_in[2 + 4 * l];
    const float* gamma = (const float*)d_in[3 + 4 * l];
    const float* beta  = (const float*)d_in[4 + 4 * l];
    const int Din = (l == 0) ? 240 : GDIM;
    const int Kp  = (l == 0) ? 256 : GDIM;

    ligru_conv_w<<<GDIM, 256, 0, stream>>>(W, Wb, Din, Kp);
    ligru_conv_u<<<GDIM, 256, 0, stream>>>(U, Ub);
    ligru_wx<<<dim3(T_LEN, 8), 256, 0, stream>>>(cur, Wb, gamma, beta, wbuf, Kp);
    ligru_recurrence<<<1, 1024, 0, stream>>>(wbuf, Ub, nxt, (float*)d_out,
                                             (l == 3) ? 1 : 0);
    bf16* tmp = cur; cur = nxt; nxt = tmp;
  }
}
